// TriangleMultiplicativeUpdate_68384469286923
// MI455X (gfx1250) — matrix operand order repaired and checked by reading, not compiled, not run on hardware
//
#include <hip/hip_runtime.h>
#include <hip/hip_bf16.h>
#include <math.h>

// ---------------------------------------------------------------------------
// Triangle multiplicative update (outgoing), N=768, D=128, fp32 in/out.
// Matmuls on v_wmma_f32_16x16x32_bf16; einsum B-panel staged to LDS via
// global_load_async_to_lds_b128 (ASYNCcnt) once per block.
// ---------------------------------------------------------------------------

typedef __bf16 bf16_t;
typedef __attribute__((ext_vector_type(16))) __bf16 v16bf;
typedef __attribute__((ext_vector_type(8)))  __bf16 v8bf;
typedef __attribute__((ext_vector_type(8)))  float  v8f;

constexpr int   Nn  = 768;
constexpr int   Dd  = 128;
constexpr int   NN  = Nn * Nn;                 // 589824
constexpr size_t NND = (size_t)NN * Dd;        // 75,497,472 elements
constexpr float EPS = 1e-5f;

// Einsum LDS B-panel: 64 rows x 768 K, rows padded to 776 bf16 (16B-aligned,
// breaks the 1536B row stride that would put all 16 lanes on one bank).
constexpr int PADK = 776;
constexpr int SMEM_B_BYTES = 64 * PADK * 2;    // 99,328 B < 320 KB/WGP

__device__ inline v8f zero8() {
  v8f z = {0.f,0.f,0.f,0.f,0.f,0.f,0.f,0.f};
  return z;
}

__device__ inline float wave_sum32(float x) {
  #pragma unroll
  for (int m = 16; m >= 1; m >>= 1) x += __shfl_xor(x, m, 32);
  return x;
}

__device__ inline float sigmoidf(float x) {
  return 1.0f / (1.0f + __expf(-x));
}

// A-fragment (16x32 bf16, MxK) from a row-major bf16 row pointer.
// Lanes 0-15 row m hold K {0..15}; lanes 16-31 row m hold K {16..31}.
__device__ inline v16bf load_a_frag(const bf16_t* row, int k0, int laneHi) {
  const bf16_t* p0 = row + k0 + laneHi * 16;
  v8bf lo = *(const v8bf*)(p0);
  v8bf hi = *(const v8bf*)(p0 + 8);
  v16bf a;
  #pragma unroll
  for (int i = 0; i < 8; ++i) { a[i] = lo[i]; a[i + 8] = hi[i]; }
  return a;
}

// B-fragment (32x16 bf16, KxN): source row n = B column n, contiguous in K.
// Lanes 0-15: col n, K=k0..k0+15; lanes 16-31: col n, K=k0+16..k0+31.
__device__ inline v16bf load_b_frag(const bf16_t* colrow, int k0, int laneHi) {
  const bf16_t* p = colrow + k0 + laneHi * 16;
  v8bf lo = *(const v8bf*)(p);
  v8bf hi = *(const v8bf*)(p + 8);
  v16bf b;
  #pragma unroll
  for (int i = 0; i < 8; ++i) { b[i] = lo[i]; b[i + 8] = hi[i]; }
  return b;
}

// B-fragment from an fp32 [out,in] weight matrix (torch Linear layout):
// B[k][o] = W[o][k]; lane = output channel o, 16 contiguous fp32 converted.
__device__ inline v16bf load_w_frag(const float* W, int oBase, int k0, int ln, int laneHi) {
  const float4* wr = (const float4*)(W + (oBase + ln) * Dd + k0 + laneHi * 16);
  v16bf b;
  #pragma unroll
  for (int q = 0; q < 4; ++q) {
    float4 f = wr[q];
    b[4*q+0] = (__bf16)f.x; b[4*q+1] = (__bf16)f.y;
    b[4*q+2] = (__bf16)f.z; b[4*q+3] = (__bf16)f.w;
  }
  return b;
}

// One per-lane 16B async global->LDS copy (CDNA5 VGLOBAL async op, ASYNCcnt).
__device__ inline void async_copy_b128(const bf16_t* gsrc, bf16_t* ldst) {
  uint32_t laddr = (uint32_t)(uintptr_t)ldst;   // LDS aperture: addr[31:0] = LDS offset
  uint64_t gaddr = (uint64_t)(uintptr_t)gsrc;
  asm volatile("global_load_async_to_lds_b128 %0, %1, off"
               :: "v"(laddr), "v"(gaddr) : "memory");
}

__device__ inline void wait_async_all() {
  asm volatile("s_wait_asynccnt 0x0" ::: "memory");
}

// ---------------------------------------------------------------------------
// K1: z[r][d] = bf16(LayerNorm(pair[r])) ; one wave per row, 8 rows/block
// ---------------------------------------------------------------------------
__global__ __launch_bounds__(256) void ln_in_kernel(
    const float* __restrict__ pair, const float* __restrict__ w,
    const float* __restrict__ b, bf16_t* __restrict__ z) {
  int wave = threadIdx.x >> 5, lane = threadIdx.x & 31;
  int r = blockIdx.x * 8 + wave;
  const float* row = pair + (size_t)r * Dd;
  float x[4], s = 0.f;
  #pragma unroll
  for (int q = 0; q < 4; ++q) { x[q] = row[lane + 32 * q]; s += x[q]; }
  float mean = wave_sum32(s) * (1.f / Dd);
  float v = 0.f;
  #pragma unroll
  for (int q = 0; q < 4; ++q) { float d = x[q] - mean; v += d * d; }
  float rstd = rsqrtf(wave_sum32(v) * (1.f / Dd) + EPS);
  bf16_t* zr = z + (size_t)r * Dd;
  #pragma unroll
  for (int q = 0; q < 4; ++q) {
    int d = lane + 32 * q;
    zr[d] = (__bf16)((x[q] - mean) * rstd * w[d] + b[d]);
  }
}

// ---------------------------------------------------------------------------
// K2: left/right projections + sigmoid gates via WMMA.
// Output stored CHANNEL-MAJOR: dst[d * NN + r]  (r = i*N + k flattened).
// ---------------------------------------------------------------------------
__global__ __launch_bounds__(256) void proj_gate_kernel(
    const bf16_t* __restrict__ z,
    const float* __restrict__ lg_w, const float* __restrict__ lg_b,
    const float* __restrict__ lp_w, const float* __restrict__ lp_b,
    const float* __restrict__ rg_w, const float* __restrict__ rg_b,
    const float* __restrict__ rp_w, const float* __restrict__ rp_b,
    bf16_t* __restrict__ left_t, bf16_t* __restrict__ right_t) {
  int wave = threadIdx.x >> 5, lane = threadIdx.x & 31;
  int ln = lane & 15, laneHi = lane >> 4;
  int rBase = blockIdx.x * 128 + wave * 16;

  v16bf afrag[4];
  const bf16_t* arow = z + (size_t)(rBase + ln) * Dd;
  #pragma unroll
  for (int s = 0; s < 4; ++s) afrag[s] = load_a_frag(arow, 32 * s, laneHi);

  #pragma unroll
  for (int side = 0; side < 2; ++side) {
    const float* Wg = side ? rg_w : lg_w;
    const float* Bg = side ? rg_b : lg_b;
    const float* Wp = side ? rp_w : lp_w;
    const float* Bp = side ? rp_b : lp_b;
    bf16_t* dst = side ? right_t : left_t;
    #pragma unroll
    for (int ot = 0; ot < 8; ++ot) {
      int oBase = ot * 16;
      v8f cg = zero8(), cp = zero8();
      #pragma unroll
      for (int s = 0; s < 4; ++s) {
        v16bf bg = load_w_frag(Wg, oBase, 32 * s, ln, laneHi);
        cg = __builtin_amdgcn_wmma_f32_16x16x32_bf16(false, afrag[s], false, bg,
                                                     (short)0, cg, false, false);
        v16bf bp = load_w_frag(Wp, oBase, 32 * s, ln, laneHi);
        cp = __builtin_amdgcn_wmma_f32_16x16x32_bf16(false, afrag[s], false, bp,
                                                     (short)0, cp, false, false);
      }
      float biasg = Bg[oBase + ln];
      float biasp = Bp[oBase + ln];
      bf16_t* dcol = dst + (size_t)(oBase + ln) * NN + rBase + 8 * laneHi;
      #pragma unroll
      for (int v = 0; v < 8; ++v) {
        float g = sigmoidf(cg[v] + biasg);
        dcol[v] = (__bf16)(g * (cp[v] + biasp));
      }
    }
  }
}

// ---------------------------------------------------------------------------
// K3: tri[i,j,d] = sum_k left[i,k,d] * right[j,k,d]
// Per channel d: C_d = L_d(768x768) @ R_d(768x768)^T (L2-resident, 2.4 MB).
// Block: 256(M) x 64(J) per channel; wave: 32x64 (8 accumulators).
// B panel (64 x 768 bf16, 96 KB) staged to LDS once via async loads; whole
// K-loop then reads B from LDS (ds_load_b128) and A from L2.
// tri stored channel-major: tri[d*NN + i*N + j].
// ---------------------------------------------------------------------------
__global__ __launch_bounds__(256) void tri_einsum_kernel(
    const bf16_t* __restrict__ left_t, const bf16_t* __restrict__ right_t,
    bf16_t* __restrict__ tri) {
  extern __shared__ bf16_t smemB[];   // [64][PADK]
  int tid = threadIdx.x;
  int wave = tid >> 5, lane = tid & 31;
  int ln = lane & 15, laneHi = lane >> 4;
  int d = blockIdx.z;
  int mBase = blockIdx.x * 256 + wave * 32;
  int jBase = blockIdx.y * 64;
  const bf16_t* L = left_t + (size_t)d * NN;
  const bf16_t* Rpanel = right_t + (size_t)d * NN + (size_t)jBase * Nn;

  // ---- Stage B panel: 64 rows x 768 cols = 6144 x 16B chunks, 24 per thread.
  #pragma unroll
  for (int it = 0; it < 24; ++it) {
    int c = tid + it * 256;
    int row = c / 96;                 // 96 chunks of 16B per 1536B row
    int col = (c % 96) * 8;           // element offset (8 bf16 = 16B)
    async_copy_b128(Rpanel + row * Nn + col, &smemB[row * PADK + col]);
  }
  wait_async_all();
  __syncthreads();

  // ---- Main K loop: 24 steps of K=32; 8 WMMAs per step.
  v8f acc[8];
  #pragma unroll
  for (int t = 0; t < 8; ++t) acc[t] = zero8();

  const bf16_t* arow0 = L + (size_t)(mBase + ln) * Nn;
  const bf16_t* arow1 = arow0 + (size_t)16 * Nn;

  for (int k0 = 0; k0 < Nn; k0 += 32) {
    v16bf a0 = load_a_frag(arow0, k0, laneHi);
    v16bf a1 = load_a_frag(arow1, k0, laneHi);
    #pragma unroll
    for (int t = 0; t < 4; ++t) {
      v16bf b = load_b_frag(&smemB[(t * 16 + ln) * PADK], k0, laneHi);
      acc[t]     = __builtin_amdgcn_wmma_f32_16x16x32_bf16(false, a0, false, b,
                                                           (short)0, acc[t], false, false);
      acc[4 + t] = __builtin_amdgcn_wmma_f32_16x16x32_bf16(false, a1, false, b,
                                                           (short)0, acc[4 + t], false, false);
    }
  }

  bf16_t* crow = tri + (size_t)d * NN;
  #pragma unroll
  for (int h = 0; h < 2; ++h) {
    #pragma unroll
    for (int t = 0; t < 4; ++t) {
      int j = jBase + t * 16 + ln;
      #pragma unroll
      for (int v = 0; v < 8; ++v) {
        int i = mBase + 16 * h + v + 8 * laneHi;
        crow[(size_t)i * Nn + j] = (__bf16)acc[4 * h + t][v];
      }
    }
  }
}

// ---------------------------------------------------------------------------
// K4a: LN over channels of tri (channel-major reads, coalesced across threads)
// -> t2 row-major bf16; also pair -> bf16 for the gate GEMM.
// ---------------------------------------------------------------------------
__global__ __launch_bounds__(256) void cn_ln_kernel(
    const bf16_t* __restrict__ tri, const float* __restrict__ cn_w,
    const float* __restrict__ cn_b, const float* __restrict__ pair,
    bf16_t* __restrict__ t2, bf16_t* __restrict__ pb) {
  int r = blockIdx.x * 256 + threadIdx.x;
  float s = 0.f, ss = 0.f;
  for (int d = 0; d < Dd; ++d) {
    float x = (float)tri[(size_t)d * NN + r];
    s += x; ss += x * x;
  }
  float mean = s * (1.f / Dd);
  float var = ss * (1.f / Dd) - mean * mean;
  float rstd = rsqrtf(var + EPS);
  bf16_t* trow = t2 + (size_t)r * Dd;
  bf16_t* prow = pb + (size_t)r * Dd;
  const float* pr = pair + (size_t)r * Dd;
  for (int d = 0; d < Dd; ++d) {
    float x = (float)tri[(size_t)d * NN + r];
    trow[d] = (__bf16)((x - mean) * rstd * cn_w[d] + cn_b[d]);
    prow[d] = (__bf16)pr[d];
  }
}

// ---------------------------------------------------------------------------
// K4b: u = sigmoid(pair @ og_w^T + og_b) * (LN(tri) @ op_w^T + op_b)  via WMMA
// u stored row-major bf16.
// ---------------------------------------------------------------------------
__global__ __launch_bounds__(256) void out_proj_kernel(
    const bf16_t* __restrict__ t2, const bf16_t* __restrict__ pb,
    const float* __restrict__ op_w, const float* __restrict__ op_b,
    const float* __restrict__ og_w, const float* __restrict__ og_b,
    bf16_t* __restrict__ u) {
  int wave = threadIdx.x >> 5, lane = threadIdx.x & 31;
  int ln = lane & 15, laneHi = lane >> 4;
  int rBase = blockIdx.x * 128 + wave * 16;

  v16bf ap[4], ag[4];
  const bf16_t* trow = t2 + (size_t)(rBase + ln) * Dd;
  const bf16_t* prow = pb + (size_t)(rBase + ln) * Dd;
  #pragma unroll
  for (int s = 0; s < 4; ++s) {
    ap[s] = load_a_frag(trow, 32 * s, laneHi);
    ag[s] = load_a_frag(prow, 32 * s, laneHi);
  }

  #pragma unroll
  for (int ot = 0; ot < 8; ++ot) {
    int oBase = ot * 16;
    v8f cg = zero8(), cp = zero8();
    #pragma unroll
    for (int s = 0; s < 4; ++s) {
      v16bf bg = load_w_frag(og_w, oBase, 32 * s, ln, laneHi);
      cg = __builtin_amdgcn_wmma_f32_16x16x32_bf16(false, ag[s], false, bg,
                                                   (short)0, cg, false, false);
      v16bf bp = load_w_frag(op_w, oBase, 32 * s, ln, laneHi);
      cp = __builtin_amdgcn_wmma_f32_16x16x32_bf16(false, ap[s], false, bp,
                                                   (short)0, cp, false, false);
    }
    float biasg = og_b[oBase + ln];
    float biasp = op_b[oBase + ln];
    #pragma unroll
    for (int v = 0; v < 8; ++v) {
      float g = sigmoidf(cg[v] + biasg);
      u[(size_t)(rBase + v + 8 * laneHi) * Dd + oBase + ln] =
          (__bf16)(g * (cp[v] + biasp));
    }
  }
}

// ---------------------------------------------------------------------------
// K5: out = LayerNorm(pair + u)  (fp32 out)
// ---------------------------------------------------------------------------
__global__ __launch_bounds__(256) void ln_out_kernel(
    const float* __restrict__ pair, const bf16_t* __restrict__ u,
    const float* __restrict__ w, const float* __restrict__ b,
    float* __restrict__ out) {
  int wave = threadIdx.x >> 5, lane = threadIdx.x & 31;
  int r = blockIdx.x * 8 + wave;
  const float* prow = pair + (size_t)r * Dd;
  const bf16_t* urow = u + (size_t)r * Dd;
  float x[4], s = 0.f;
  #pragma unroll
  for (int q = 0; q < 4; ++q) {
    int d = lane + 32 * q;
    x[q] = prow[d] + (float)urow[d];
    s += x[q];
  }
  float mean = wave_sum32(s) * (1.f / Dd);
  float v = 0.f;
  #pragma unroll
  for (int q = 0; q < 4; ++q) { float t = x[q] - mean; v += t * t; }
  float rstd = rsqrtf(wave_sum32(v) * (1.f / Dd) + EPS);
  float* orow = out + (size_t)r * Dd;
  #pragma unroll
  for (int q = 0; q < 4; ++q) {
    int d = lane + 32 * q;
    orow[d] = (x[q] - mean) * rstd * w[d] + b[d];
  }
}

// ---------------------------------------------------------------------------
extern "C" void kernel_launch(void* const* d_in, const int* in_sizes, int n_in,
                              void* d_out, int out_size, void* d_ws, size_t ws_size,
                              hipStream_t stream) {
  const float* pair    = (const float*)d_in[0];
  const float* ln_in_w = (const float*)d_in[1];
  const float* ln_in_b = (const float*)d_in[2];
  const float* lp_w    = (const float*)d_in[3];
  const float* lp_b    = (const float*)d_in[4];
  const float* lg_w    = (const float*)d_in[5];
  const float* lg_b    = (const float*)d_in[6];
  const float* rp_w    = (const float*)d_in[7];
  const float* rp_b    = (const float*)d_in[8];
  const float* rg_w    = (const float*)d_in[9];
  const float* rg_b    = (const float*)d_in[10];
  const float* cn_w    = (const float*)d_in[11];
  const float* cn_b    = (const float*)d_in[12];
  const float* op_w    = (const float*)d_in[13];
  const float* op_b    = (const float*)d_in[14];
  const float* og_w    = (const float*)d_in[15];
  const float* og_b    = (const float*)d_in[16];
  const float* ln_out_w = (const float*)d_in[17];
  const float* ln_out_b = (const float*)d_in[18];
  float* out = (float*)d_out;

  // Workspace: 3 regions of NND bf16 (151 MB each), with dead-buffer aliasing.
  bf16_t* z       = (bf16_t*)d_ws;       // K1 out; dead after K2
  bf16_t* left_t  = z + NND;             // K2 out; dead after K3
  bf16_t* right_t = left_t + NND;        // K2 out; dead after K3
  bf16_t* tri     = z;                   // K3 out (aliases z); dead after K4a
  bf16_t* t2      = left_t;              // K4a out (aliases left_t)
  bf16_t* pb      = right_t;             // K4a out (aliases right_t)
  bf16_t* u       = z;                   // K4b out (aliases tri region)

  ln_in_kernel<<<NN / 8, 256, 0, stream>>>(pair, ln_in_w, ln_in_b, z);
  proj_gate_kernel<<<NN / 128, 256, 0, stream>>>(z, lg_w, lg_b, lp_w, lp_b,
                                                 rg_w, rg_b, rp_w, rp_b,
                                                 left_t, right_t);
  tri_einsum_kernel<<<dim3(Nn / 256, Nn / 64, Dd), 256, SMEM_B_BYTES, stream>>>(
      left_t, right_t, tri);
  cn_ln_kernel<<<NN / 256, 256, 0, stream>>>(tri, cn_w, cn_b, pair, t2, pb);
  out_proj_kernel<<<NN / 128, 256, 0, stream>>>(t2, pb, op_w, op_b, og_w, og_b, u);
  ln_out_kernel<<<NN / 8, 256, 0, stream>>>(pair, u, ln_out_w, ln_out_b, out);
}
